// GraphAttention_1374389534738
// MI455X (gfx1250) — matrix operand order repaired and checked by reading, not compiled, not run on hardware
//
#include <hip/hip_runtime.h>
#include <hip/hip_bf16.h>

typedef __attribute__((ext_vector_type(16))) __bf16 v16bf;
typedef __attribute__((ext_vector_type(8)))  __bf16 v8bf;
typedef __attribute__((ext_vector_type(8)))  float  v8f;
typedef __attribute__((ext_vector_type(4)))  float  v4f;
typedef __attribute__((ext_vector_type(4)))  unsigned v4u;
typedef __attribute__((ext_vector_type(8)))  int    v8i;
typedef __attribute__((ext_vector_type(4)))  int    v4i;

#define DM 512
#define NH 8
#define DH 64
#define BB 8
#define NN 1024
#define MTOT (BB * NN)          // 8192
#define WTILE_B 5248            // padded 64x(32+8) bf16 TDM tile + slack, bytes

__device__ __forceinline__ __bf16 f2bf(float f) { return (__bf16)f; }

__device__ __forceinline__ v16bf ld16bf(const __bf16* p) {
    v8bf lo = *(const v8bf*)p;
    v8bf hi = *(const v8bf*)(p + 8);
    v16bf r;
#pragma unroll
    for (int i = 0; i < 8; ++i) { r[i] = lo[i]; r[8 + i] = hi[i]; }
    return r;
}

// ---------------------------------------------------------------------------
// TDM: load one 64-col x 32-k bf16 weight tile into LDS, padded so each
// 32-element row lands on a 40-element (80 B) stride (pad 4 DWORDs per 16).
// 2D descriptor: data_size=2B, tile_dim0=32 (k, contiguous), tile_dim1=64
// (cols), tensor_dim0_stride=512.
// ---------------------------------------------------------------------------
__device__ __forceinline__ void tdm_load_wtile(const __bf16* gsrc, unsigned lds_off) {
    unsigned long long ga = (unsigned long long)(size_t)gsrc;
    v4u g0;
    g0[0] = 1u;                                   // count=1 (valid user D#)
    g0[1] = lds_off;                              // lds_addr (bytes)
    g0[2] = (unsigned)ga;                         // global_addr[31:0]
    g0[3] = (unsigned)((ga >> 32) & 0x1FFFFFFull) | (2u << 30); // addr[56:32]|type=2
    v8i g1;
    g1[0] = (int)((1u << 16) | (1u << 20) | (3u << 22) | (3u << 25));
        // data_size=2B | pad_enable | pad_interval=16DW | pad_amount=4DW
    g1[1] = (int)(512u << 16);                    // tensor_dim0 = 512 -> bits[79:48]
    g1[2] = (int)(4096u << 16);                   // tensor_dim1 = 4096 -> bits[111:80]
    g1[3] = (int)(32u << 16);                     // tile_dim0 = 32 -> bits[127:112]
    g1[4] = (int)64;                              // tile_dim1 = 64 -> bits[143:128]
    g1[5] = (int)512;                             // tensor_dim0_stride = 512
    g1[6] = 0; g1[7] = 0;
    v4i z = {0, 0, 0, 0};
#if __clang_major__ >= 23
    v8i z8 = {0, 0, 0, 0, 0, 0, 0, 0};
    __builtin_amdgcn_tensor_load_to_lds(g0, g1, z, z, z8, 0);
#else
    __builtin_amdgcn_tensor_load_to_lds(g0, g1, z, z, 0);
#endif
}

// ---------------------------------------------------------------------------
// f32 -> bf16 bulk convert (x and the four weight matrices), 8 elems/thread.
// ---------------------------------------------------------------------------
__global__ __launch_bounds__(256) void cvt_kernel(
    const float* __restrict__ src, __bf16* __restrict__ dst, int n)
{
    const int i = (blockIdx.x * 256 + threadIdx.x) * 8;
    if (i >= n) return;
    v4f a = *(const v4f*)(src + i);
    v4f b = *(const v4f*)(src + i + 4);
#pragma unroll
    for (int j = 0; j < 4; ++j) { dst[i + j] = f2bf(a[j]); dst[i + 4 + j] = f2bf(b[j]); }
}

// ---------------------------------------------------------------------------
// GEMM  Y[m,c] = sum_k A[m,k]*Wb[c,k] (+bias) ; A,Wb bf16. M=8192,N=512,K=512.
// 256 thr = 8 waves stacked in M; block tile 128M x 64N; wave tile 16M x 64N.
// W tiles arrive via TDM (double-buffered LDS).
// mode 0: bf16 out [M,512] ; mode 1: bf16 out transposed [B][512][1024]
// mode 2: f32 out H = acc + bias + X   (output projection + residual)
// ---------------------------------------------------------------------------
__global__ __launch_bounds__(256) void gemm_kernel(
    const __bf16* __restrict__ A, const __bf16* __restrict__ Wb,
    const float* __restrict__ bias, const float* __restrict__ X,
    __bf16* __restrict__ Ybf, float* __restrict__ Yf, int mode)
{
    __shared__ __bf16 wlds[2 * (WTILE_B / 2)];
    const int tid   = threadIdx.x;
    const int lane  = tid & 31;
    const int wave  = tid >> 5;
    const int l15   = lane & 15;
    const int lhalf = lane >> 4;
    const int m0    = blockIdx.x * 128 + wave * 16;
    const int c0    = blockIdx.y * 64;
    const unsigned lds0 = (unsigned)(size_t)(&wlds[0]);

    if (wave == 0) tdm_load_wtile(Wb + (size_t)c0 * DM, lds0);

    v8f acc[4] = {};
    int cur = 0;
    for (int kb = 0; kb < DM; kb += 32) {
        if (wave == 0) __builtin_amdgcn_s_wait_tensorcnt(0);
        __syncthreads();
        if (wave == 0 && kb + 32 < DM)
            tdm_load_wtile(Wb + (size_t)c0 * DM + kb + 32, lds0 + (cur ^ 1) * WTILE_B);

        // A fragment (16x32 bf16): lane row m0+l15, K [16*lhalf,+16)
        v16bf afr = ld16bf(A + (size_t)(m0 + l15) * DM + kb + lhalf * 16);
        const __bf16* wbuf = wlds + cur * (WTILE_B / 2);
        v16bf bfr0 = ld16bf(wbuf + (0 * 16 + l15) * 40 + lhalf * 16);
        v16bf bfr1 = ld16bf(wbuf + (1 * 16 + l15) * 40 + lhalf * 16);
        v16bf bfr2 = ld16bf(wbuf + (2 * 16 + l15) * 40 + lhalf * 16);
        v16bf bfr3 = ld16bf(wbuf + (3 * 16 + l15) * 40 + lhalf * 16);
        acc[0] = __builtin_amdgcn_wmma_f32_16x16x32_bf16(false, afr, false, bfr0, (short)0, acc[0], false, false);
        acc[1] = __builtin_amdgcn_wmma_f32_16x16x32_bf16(false, afr, false, bfr1, (short)0, acc[1], false, false);
        acc[2] = __builtin_amdgcn_wmma_f32_16x16x32_bf16(false, afr, false, bfr2, (short)0, acc[2], false, false);
        acc[3] = __builtin_amdgcn_wmma_f32_16x16x32_bf16(false, afr, false, bfr3, (short)0, acc[3], false, false);
        cur ^= 1;
    }

    const int b = m0 >> 10;
#pragma unroll
    for (int ct = 0; ct < 4; ++ct) {
        const int c  = c0 + ct * 16 + l15;
        const float bv = bias[c];
        if (mode == 0) {
#pragma unroll
            for (int j = 0; j < 8; ++j) {
                const int m = m0 + lhalf * 8 + j;
                Ybf[(size_t)m * DM + c] = f2bf(acc[ct][j] + bv);
            }
        } else if (mode == 1) {
            const int nrow = (m0 & (NN - 1)) + lhalf * 8;
            __bf16* yp = Ybf + (size_t)b * DM * NN + (size_t)c * NN + nrow;
#pragma unroll
            for (int j = 0; j < 8; ++j) yp[j] = f2bf(acc[ct][j] + bv);
        } else {
#pragma unroll
            for (int j = 0; j < 8; ++j) {
                const int m = m0 + lhalf * 8 + j;
                Yf[(size_t)m * DM + c] = acc[ct][j] + bv + X[(size_t)m * DM + c];
            }
        }
    }
}

// ---------------------------------------------------------------------------
// Flash-style masked attention. Grid (N/64, H, B), 128 thr = 4 waves,
// one 16-query tile per wave. Streams 32-key blocks; online softmax.
// Qb,Kb: [B,N,512] bf16 ; Vt: [B,512,1024] bf16 ; Ao: [B,N,512] bf16.
// ---------------------------------------------------------------------------
__global__ __launch_bounds__(128) void attn_kernel(
    const __bf16* __restrict__ Qb, const __bf16* __restrict__ Kb,
    const __bf16* __restrict__ Vt, const float* __restrict__ adj,
    __bf16* __restrict__ Ao)
{
    __shared__ __bf16 plds[4 * 16 * 40];     // wave-private 16x32 P tiles
    const int lane  = threadIdx.x & 31;
    const int wave  = threadIdx.x >> 5;
    const int l15   = lane & 15;
    const int lhalf = lane >> 4;
    const int b = blockIdx.z, h = blockIdx.y;
    const int q0 = blockIdx.x * 64 + wave * 16;

    // Q fragments for d-windows [0,32) and [32,64)
    v16bf qa[2];
    {
        const __bf16* qrow = Qb + ((size_t)b * NN + q0 + l15) * DM + h * DH;
        qa[0] = ld16bf(qrow + lhalf * 16);
        qa[1] = ld16bf(qrow + 32 + lhalf * 16);
    }

    float mrow[8], lrow[8];
#pragma unroll
    for (int j = 0; j < 8; ++j) { mrow[j] = -1e30f; lrow[j] = 0.f; }
    v8f acc[4] = {};

    const __bf16* krow  = Kb + (size_t)b * NN * DM + h * DH;              // + key*512
    const __bf16* vbase = Vt + (size_t)b * DM * NN + (size_t)h * DH * NN; // + d*1024
    __bf16* pl = plds + wave * 16 * 40;

    for (int kb = 0; kb < NN; kb += 32) {
        if (kb + 32 < NN) {   // prefetch next key block (K rows + V columns)
            __builtin_prefetch(krow + (size_t)(kb + 32 + lane) * DM, 0, 1);
            __builtin_prefetch(vbase + (size_t)(lane * 2) * NN + kb + 32, 0, 1);
        }
        // scores: two 16x16 tiles (keys kb..kb+15, kb+16..kb+31)
        v8f s0 = {}, s1 = {};
#pragma unroll
        for (int dd = 0; dd < 2; ++dd) {
            v16bf k0 = ld16bf(krow + (size_t)(kb + l15)      * DM + dd * 32 + lhalf * 16);
            v16bf k1 = ld16bf(krow + (size_t)(kb + 16 + l15) * DM + dd * 32 + lhalf * 16);
            s0 = __builtin_amdgcn_wmma_f32_16x16x32_bf16(false, qa[dd], false, k0, (short)0, s0, false, false);
            s1 = __builtin_amdgcn_wmma_f32_16x16x32_bf16(false, qa[dd], false, k1, (short)0, s1, false, false);
        }

        // mask (adj + self-loop), online softmax update per owned row
        const int k0i = kb + l15, k1i = kb + 16 + l15;
#pragma unroll
        for (int j = 0; j < 8; ++j) {
            const int q = q0 + lhalf * 8 + j;
            const float a0 = adj[((size_t)b * NN + q) * NN + k0i];
            const float a1 = adj[((size_t)b * NN + q) * NN + k1i];
            float v0 = s0[j] * 0.125f, v1 = s1[j] * 0.125f;
            if (!(a0 > 0.f || q == k0i)) v0 = -1e30f;
            if (!(a1 > 0.f || q == k1i)) v1 = -1e30f;
            float loc = fmaxf(v0, v1);
            loc = fmaxf(loc, __shfl_xor(loc, 1, 32));
            loc = fmaxf(loc, __shfl_xor(loc, 2, 32));
            loc = fmaxf(loc, __shfl_xor(loc, 4, 32));
            loc = fmaxf(loc, __shfl_xor(loc, 8, 32));
            const float mn    = fmaxf(mrow[j], loc);
            const float alpha = __expf(mrow[j] - mn);
            const float p0 = __expf(v0 - mn), p1 = __expf(v1 - mn);
            float rs = p0 + p1;
            rs += __shfl_xor(rs, 1, 32);
            rs += __shfl_xor(rs, 2, 32);
            rs += __shfl_xor(rs, 4, 32);
            rs += __shfl_xor(rs, 8, 32);
            lrow[j] = lrow[j] * alpha + rs;
            mrow[j] = mn;
#pragma unroll
            for (int t = 0; t < 4; ++t) acc[t][j] *= alpha;
            s0[j] = p0; s1[j] = p1;
        }

        // P: C-layout -> A-fragment layout via wave-private LDS bounce.
        // Same-wave DS ops are pipeline-ordered; s_wait_dscnt 0 + compiler
        // fence is enough (no workgroup barrier needed).
#pragma unroll
        for (int j = 0; j < 8; ++j) {
            const int r = lhalf * 8 + j;
            pl[r * 40 + l15]      = f2bf(s0[j]);
            pl[r * 40 + 16 + l15] = f2bf(s1[j]);
        }
        asm volatile("s_wait_dscnt 0x0" ::: "memory");
        v16bf pa;
        {
            const __bf16* pp = pl + l15 * 40 + lhalf * 16;
            v8bf c0v = *(const v8bf*)pp;
            v8bf c1v = *(const v8bf*)(pp + 8);
#pragma unroll
            for (int i = 0; i < 8; ++i) { pa[i] = c0v[i]; pa[8 + i] = c1v[i]; }
        }
        asm volatile("" ::: "memory");   // keep loads before next iter's stores

        // O += P @ V  (4 d-tiles of 16)
#pragma unroll
        for (int t = 0; t < 4; ++t) {
            v16bf vfr = ld16bf(vbase + (size_t)(t * 16 + l15) * NN + kb + lhalf * 16);
            acc[t] = __builtin_amdgcn_wmma_f32_16x16x32_bf16(
                false, pa, false, vfr, (short)0, acc[t], false, false);
        }
    }

    // finalize: divide by row sums, store merged-head bf16 output
#pragma unroll
    for (int j = 0; j < 8; ++j) {
        const int q = q0 + lhalf * 8 + j;
        const float inv = 1.f / lrow[j];
#pragma unroll
        for (int t = 0; t < 4; ++t)
            Ao[((size_t)b * NN + q) * DM + h * DH + t * 16 + l15] = f2bf(acc[t][j] * inv);
    }
}

// ---------------------------------------------------------------------------
// LayerNorm over last dim (512). One wave per row, shfl reductions.
// ---------------------------------------------------------------------------
__global__ __launch_bounds__(256) void ln_kernel(
    const float* __restrict__ H, const float* __restrict__ gamma,
    const float* __restrict__ beta, float* __restrict__ out)
{
    const int lane = threadIdx.x & 31;
    const int wave = threadIdx.x >> 5;
    const int row  = blockIdx.x * 8 + wave;
    const float* hr = H + (size_t)row * DM;

    float v[16], s = 0.f;
#pragma unroll
    for (int i = 0; i < 16; ++i) { v[i] = hr[lane + i * 32]; s += v[i]; }
#pragma unroll
    for (int m = 1; m < 32; m <<= 1) s += __shfl_xor(s, m, 32);
    const float mu = s * (1.f / DM);
    float var = 0.f;
#pragma unroll
    for (int i = 0; i < 16; ++i) { const float d = v[i] - mu; var += d * d; }
#pragma unroll
    for (int m = 1; m < 32; m <<= 1) var += __shfl_xor(var, m, 32);
    const float r = rsqrtf(var * (1.f / DM) + 1e-5f);
    float* orow = out + (size_t)row * DM;
#pragma unroll
    for (int i = 0; i < 16; ++i) {
        const int c = lane + i * 32;
        orow[c] = (v[i] - mu) * r * gamma[c] + beta[c];
    }
}

extern "C" void kernel_launch(void* const* d_in, const int* in_sizes, int n_in,
                              void* d_out, int out_size, void* d_ws, size_t ws_size,
                              hipStream_t stream) {
    const float* x     = (const float*)d_in[0];
    const float* adj   = (const float*)d_in[1];
    const float* Wq    = (const float*)d_in[2];
    const float* bq    = (const float*)d_in[3];
    const float* Wk    = (const float*)d_in[4];
    const float* bk    = (const float*)d_in[5];
    const float* Wv    = (const float*)d_in[6];
    const float* bv    = (const float*)d_in[7];
    const float* Wo    = (const float*)d_in[8];
    const float* bo    = (const float*)d_in[9];
    const float* gamma = (const float*)d_in[10];
    const float* beta  = (const float*)d_in[11];

    char* ws = (char*)d_ws;
    __bf16* Xbf = (__bf16*)(ws);                              // 8 MB
    __bf16* Qb  = (__bf16*)(ws + (size_t)( 8 << 20));         // 8 MB
    __bf16* Kb2 = (__bf16*)(ws + (size_t)(16 << 20));         // 8 MB
    __bf16* Vt  = (__bf16*)(ws + (size_t)(24 << 20));         // 8 MB (transposed)
    __bf16* Ao  = (__bf16*)(ws + (size_t)(32 << 20));         // 8 MB
    __bf16* Wqb = (__bf16*)(ws + (size_t)(40 << 20));         // 0.5 MB each
    __bf16* Wkb = (__bf16*)(ws + (size_t)(40 << 20) + (512 << 10));
    __bf16* Wvb = (__bf16*)(ws + (size_t)(41 << 20));
    __bf16* Wob = (__bf16*)(ws + (size_t)(41 << 20) + (512 << 10));
    float*  H   = (float*) (ws + (size_t)(42 << 20));         // 16 MB

    const int NX = MTOT * DM, NW = DM * DM;
    cvt_kernel<<<NX / (256 * 8), 256, 0, stream>>>(x,  Xbf, NX);
    cvt_kernel<<<NW / (256 * 8), 256, 0, stream>>>(Wq, Wqb, NW);
    cvt_kernel<<<NW / (256 * 8), 256, 0, stream>>>(Wk, Wkb, NW);
    cvt_kernel<<<NW / (256 * 8), 256, 0, stream>>>(Wv, Wvb, NW);
    cvt_kernel<<<NW / (256 * 8), 256, 0, stream>>>(Wo, Wob, NW);

    dim3 pgrid(MTOT / 128, DM / 64);                          // (64, 8)
    gemm_kernel<<<pgrid, 256, 0, stream>>>(Xbf, Wqb, bq, nullptr, Qb,  nullptr, 0);
    gemm_kernel<<<pgrid, 256, 0, stream>>>(Xbf, Wkb, bk, nullptr, Kb2, nullptr, 0);
    gemm_kernel<<<pgrid, 256, 0, stream>>>(Xbf, Wvb, bv, nullptr, Vt,  nullptr, 1);
    attn_kernel<<<dim3(NN / 64, NH, BB), 128, 0, stream>>>(Qb, Kb2, Vt, adj, Ao);
    gemm_kernel<<<pgrid, 256, 0, stream>>>(Ao, Wob, bo, x, nullptr, H, 2);
    ln_kernel<<<MTOT / 8, 256, 0, stream>>>(H, gamma, beta, (float*)d_out);
}
